// MultiHeadedAttention_82145544504022
// MI455X (gfx1250) — matrix operand order repaired and checked by reading, not compiled, not run on hardware
//
#include <hip/hip_runtime.h>

typedef __bf16 bf16_t;
typedef __attribute__((ext_vector_type(4)))  __bf16 bf16x4;
typedef __attribute__((ext_vector_type(8)))  __bf16 bf16x8;
typedef __attribute__((ext_vector_type(16))) __bf16 bf16x16;
typedef __attribute__((ext_vector_type(8)))  float  f32x8;

#define BATCH  2
#define SEQ    2048
#define DMODEL 1024
#define NHEAD  16
#define DHEAD  64
#define MTOT   (BATCH * SEQ)   // 4096

__device__ __forceinline__ bf16_t to_bf16(float x) { return (bf16_t)x; }

__device__ __forceinline__ f32x8 wmma_bf16(bf16x16 a, bf16x16 b, f32x8 c) {
  // D = A(16x32) * B(32x16) + C, f32 accumulate
  return __builtin_amdgcn_wmma_f32_16x16x32_bf16(
      /*neg_a=*/false, a, /*neg_b=*/false, b,
      /*c_mod=*/(short)0, c, /*reuse_a=*/false, /*reuse_b=*/false);
}

// A-matrix fragment (16x32, MxK), row-major source with leading dim ld.
// Lanes 0-15: row m=lane, K = 0-7 and 16-23; lanes 16-31: row m=lane-16, K = 8-15 and 24-31.
__device__ __forceinline__ bf16x16 load_frag_a(const bf16_t* p, int ld, int lane) {
  const int l15 = lane & 15;
  const int hi  = lane >> 4;
  const bf16_t* r = p + l15 * ld + hi * 8;
  bf16x8 lo = *(const bf16x8*)(r);
  bf16x8 up = *(const bf16x8*)(r + 16);
  return __builtin_shufflevector(lo, up, 0,1,2,3,4,5,6,7,8,9,10,11,12,13,14,15);
}

// B-matrix fragment (32x16, KxN). Source stored "column major" as N rows of K
// contiguous elements (leading dim ld). Lane n<16: col n, K=0-7 and 16-23; lane n+16: col n, K=8-15 and 24-31.
__device__ __forceinline__ bf16x16 load_frag_b(const bf16_t* p, int ld, int lane) {
  const int l15 = lane & 15;
  const int hi  = lane >> 4;
  const bf16_t* r = p + l15 * ld + hi * 8;
  bf16x8 lo = *(const bf16x8*)(r);
  bf16x8 up = *(const bf16x8*)(r + 16);
  return __builtin_shufflevector(lo, up, 0,1,2,3,4,5,6,7,8,9,10,11,12,13,14,15);
}

// ---------------- f32 -> bf16 conversion (vectorized x4) ----------------
__global__ void cvt4_kernel(const float* __restrict__ src, bf16_t* __restrict__ dst, int n4) {
  int i = blockIdx.x * blockDim.x + threadIdx.x;
  if (i < n4) {
    float4 f = ((const float4*)src)[i];
    bf16x4 o;
    o[0] = to_bf16(f.x); o[1] = to_bf16(f.y);
    o[2] = to_bf16(f.z); o[3] = to_bf16(f.w);
    *(bf16x4*)&dst[i * 4] = o;
  }
}

// ---------------- Projection GEMM: Y = X @ W^T + bias ----------------
// X: (MTOT x DMODEL) row-major bf16; W: (DMODEL x DMODEL) row-major bf16 (N x K).
// mode 0: outF f32 row-major (final O-projection).
// mode 1: outH bf16, (B, H, S, DK) layout (Q / K).
// mode 2: outH bf16, (B, H, DK, S) layout (V, pre-transposed for P@V).
// Double-buffered LDS, one barrier per k-step, register prefetch of next tile.
// k-loop unrolled by 2 to match buffer parity (locks in the round-2 pipelining).
__global__ __launch_bounds__(256, 1) void gemm_proj_kernel(
    const bf16_t* __restrict__ A, const bf16_t* __restrict__ W,
    const float* __restrict__ bias, float* __restrict__ outF,
    bf16_t* __restrict__ outH, int mode) {
  __shared__ bf16_t As[2][128 * 32];
  __shared__ bf16_t Bs[2][128 * 32];
  const int tid = threadIdx.x, lane = tid & 31, wave = tid >> 5;
  const int bm = blockIdx.x * 128, bn = blockIdx.y * 128;
  const int wm = (wave & 3) * 32, wn = (wave >> 2) * 64;

  const f32x8 zero = {};
  f32x8 acc[2][4];
#pragma unroll
  for (int i = 0; i < 2; ++i)
#pragma unroll
    for (int j = 0; j < 4; ++j) acc[i][j] = zero;

  const int lrow = tid >> 1;        // 0..127
  const int lcol = (tid & 1) * 16;  // 0 or 16
  const bf16_t* aptr = &A[(size_t)(bm + lrow) * DMODEL + lcol];
  const bf16_t* wptr = &W[(size_t)(bn + lrow) * DMODEL + lcol];

  // prologue prefetch of k-tile 0 into registers
  bf16x8 ra0 = *(const bf16x8*)(aptr);
  bf16x8 ra1 = *(const bf16x8*)(aptr + 8);
  bf16x8 rb0 = *(const bf16x8*)(wptr);
  bf16x8 rb1 = *(const bf16x8*)(wptr + 8);

  int buf = 0;
#pragma unroll 2
  for (int k0 = 0; k0 < DMODEL; k0 += 32) {
    *(bf16x8*)&As[buf][lrow * 32 + lcol]     = ra0;
    *(bf16x8*)&As[buf][lrow * 32 + lcol + 8] = ra1;
    *(bf16x8*)&Bs[buf][lrow * 32 + lcol]     = rb0;
    *(bf16x8*)&Bs[buf][lrow * 32 + lcol + 8] = rb1;
    __syncthreads();

    if (k0 + 32 < DMODEL) {  // prefetch next k-tile while WMMAs run
      ra0 = *(const bf16x8*)(aptr + k0 + 32);
      ra1 = *(const bf16x8*)(aptr + k0 + 40);
      rb0 = *(const bf16x8*)(wptr + k0 + 32);
      rb1 = *(const bf16x8*)(wptr + k0 + 40);
    }

    bf16x16 a0 = load_frag_a(&As[buf][(wm + 0) * 32], 32, lane);
    bf16x16 a1 = load_frag_a(&As[buf][(wm + 16) * 32], 32, lane);
#pragma unroll
    for (int j = 0; j < 4; ++j) {
      bf16x16 b = load_frag_b(&Bs[buf][(wn + j * 16) * 32], 32, lane);
      acc[0][j] = wmma_bf16(a0, b, acc[0][j]);
      acc[1][j] = wmma_bf16(a1, b, acc[1][j]);
    }
    buf ^= 1;
  }

  const int l15 = lane & 15, hi = lane >> 4;
#pragma unroll
  for (int i = 0; i < 2; ++i)
#pragma unroll
    for (int j = 0; j < 4; ++j)
#pragma unroll
      for (int e = 0; e < 8; ++e) {
        int r = bm + wm + i * 16 + e + hi * 8;
        int c = bn + wn + j * 16 + l15;
        float val = acc[i][j][e] + bias[c];
        if (mode == 0) {
          outF[(size_t)r * DMODEL + c] = val;
        } else {
          int b = r >> 11, s = r & (SEQ - 1);
          int h = c >> 6, dk = c & (DHEAD - 1);
          if (mode == 1)
            outH[(((size_t)(b * NHEAD + h) * SEQ) + s) * DHEAD + dk] = to_bf16(val);
          else  // mode 2: transposed V layout (B,H,DK,S)
            outH[(((size_t)(b * NHEAD + h) * DHEAD) + dk) * SEQ + s] = to_bf16(val);
        }
      }
}

// ---------------- Flash attention (online softmax, WMMA QK^T and PV) ----------------
// Qb/Kb: bf16 (B,H,S,DK). VtG: bf16 (B,H,DK,S). Ctx out: bf16 (B,S,D) row-major.
__global__ __launch_bounds__(128, 1) void attn_kernel(
    const bf16_t* __restrict__ Qb, const bf16_t* __restrict__ Kb,
    const bf16_t* __restrict__ VtG, const int* __restrict__ mask,
    bf16_t* __restrict__ Ctx) {
  __shared__ bf16_t Ks[64 * 64];      // [key][dk]
  __shared__ bf16_t Vts[64 * 64];     // [dk][key]  (already transposed in global)
  __shared__ bf16_t Pt[4][16 * 64];   // per-wave P staging
  const int tid = threadIdx.x, lane = tid & 31, wave = tid >> 5;
  const int qtile = blockIdx.x;       // 0..31
  const int bh = blockIdx.y;          // 0..31 = b*NHEAD + h
  const int b = bh >> 4;
  const int q0 = qtile * 64 + wave * 16;
  const int l15 = lane & 15, hi = lane >> 4;

  const bf16_t* qbase = &Qb[((size_t)bh * SEQ + q0) * DHEAD];
  bf16x16 aq0 = load_frag_a(qbase, DHEAD, lane);
  bf16x16 aq1 = load_frag_a(qbase + 32, DHEAD, lane);

  const int r2 = tid >> 1;            // 0..63 (K row / Vt row)
  const int cb = (tid & 1) * 32;      // 0 or 32
  const bf16_t* ksrc0 = &Kb[((size_t)bh * SEQ + r2) * DHEAD + cb];
  const bf16_t* vsrc0 = &VtG[((size_t)bh * DHEAD + r2) * SEQ + cb];

  const f32x8 zero = {};
  float mrow[8], lrow[8];
  f32x8 o[4];
#pragma unroll
  for (int e = 0; e < 8; ++e) { mrow[e] = -1e30f; lrow[e] = 0.f; }
#pragma unroll
  for (int j = 0; j < 4; ++j) o[j] = zero;

  for (int kb = 0; kb < SEQ; kb += 64) {
    __syncthreads();  // previous iteration's readers are done
    // cooperative vectorized loads of K block and Vt block (all b128, no scatter)
    {
      const bf16_t* ks = ksrc0 + (size_t)kb * DHEAD;
      const bf16_t* vs = vsrc0 + kb;
#pragma unroll
      for (int u = 0; u < 4; ++u) {
        *(bf16x8*)&Ks[r2 * 64 + cb + u * 8]  = *(const bf16x8*)(ks + u * 8);
        *(bf16x8*)&Vts[r2 * 64 + cb + u * 8] = *(const bf16x8*)(vs + u * 8);
      }
    }
    __syncthreads();

    if (kb + 64 < SEQ) {  // pull next block toward the caches while we compute
      __builtin_prefetch(ksrc0 + (size_t)(kb + 64) * DHEAD, 0, 3);
      __builtin_prefetch(vsrc0 + kb + 64, 0, 3);
    }

    // scores: S = (Q K^T) * scale + mask_bias
    f32x8 s[4];
#pragma unroll
    for (int j = 0; j < 4; ++j) s[j] = zero;
#pragma unroll
    for (int j = 0; j < 4; ++j) {
      bf16x16 b0 = load_frag_b(&Ks[(j * 16) * 64 + 0], 64, lane);
      bf16x16 b1 = load_frag_b(&Ks[(j * 16) * 64 + 32], 64, lane);
      s[j] = wmma_bf16(aq0, b0, s[j]);
      s[j] = wmma_bf16(aq1, b1, s[j]);
    }
#pragma unroll
    for (int j = 0; j < 4; ++j) {
      int key = kb + j * 16 + l15;
      float mb = (mask[b * SEQ + key] == 0) ? -1.0e9f : 0.0f;
#pragma unroll
      for (int e = 0; e < 8; ++e) s[j][e] = s[j][e] * 0.125f + mb;
    }

    // online softmax update (row stats across 16 lanes via shfl_xor width 16)
    float mnew[8], alpha[8];
#pragma unroll
    for (int e = 0; e < 8; ++e) {
      float v = fmaxf(fmaxf(s[0][e], s[1][e]), fmaxf(s[2][e], s[3][e]));
#pragma unroll
      for (int off = 1; off < 16; off <<= 1) v = fmaxf(v, __shfl_xor(v, off, 16));
      mnew[e] = fmaxf(mrow[e], v);
      alpha[e] = __expf(mrow[e] - mnew[e]);
    }
#pragma unroll
    for (int j = 0; j < 4; ++j)
#pragma unroll
      for (int e = 0; e < 8; ++e) s[j][e] = __expf(s[j][e] - mnew[e]);
#pragma unroll
    for (int e = 0; e < 8; ++e) {
      float v = s[0][e] + s[1][e] + s[2][e] + s[3][e];
#pragma unroll
      for (int off = 1; off < 16; off <<= 1) v += __shfl_xor(v, off, 16);
      lrow[e] = lrow[e] * alpha[e] + v;
      mrow[e] = mnew[e];
    }
#pragma unroll
    for (int j = 0; j < 4; ++j)
#pragma unroll
      for (int e = 0; e < 8; ++e) o[j][e] *= alpha[e];

    // stage P (C-layout -> LDS row-major) for A-fragment reload (wave-private)
#pragma unroll
    for (int j = 0; j < 4; ++j)
#pragma unroll
      for (int e = 0; e < 8; ++e)
        Pt[wave][(e + hi * 8) * 64 + j * 16 + l15] = to_bf16(s[j][e]);

    // O += P @ V
#pragma unroll
    for (int c = 0; c < 2; ++c) {
      bf16x16 ap = load_frag_a(&Pt[wave][c * 32], 64, lane);
#pragma unroll
      for (int j = 0; j < 4; ++j) {
        bf16x16 bv = load_frag_b(&Vts[(j * 16) * 64 + c * 32], 64, lane);
        o[j] = wmma_bf16(ap, bv, o[j]);
      }
    }
  }

  // normalize and write ctx (B,S,D) bf16
  const int h = bh & 15;
#pragma unroll
  for (int j = 0; j < 4; ++j)
#pragma unroll
    for (int e = 0; e < 8; ++e) {
      int srow = q0 + e + hi * 8;
      float v = o[j][e] / lrow[e];
      Ctx[((size_t)b * SEQ + srow) * DMODEL + h * DHEAD + j * 16 + l15] = to_bf16(v);
    }
}

extern "C" void kernel_launch(void* const* d_in, const int* in_sizes, int n_in,
                              void* d_out, int out_size, void* d_ws, size_t ws_size,
                              hipStream_t stream) {
  const float* q    = (const float*)d_in[0];
  const float* k    = (const float*)d_in[1];
  const float* v    = (const float*)d_in[2];
  const int*   mask = (const int*)d_in[3];
  const float* Wq   = (const float*)d_in[4];
  const float* bq   = (const float*)d_in[5];
  const float* Wk   = (const float*)d_in[6];
  const float* bk   = (const float*)d_in[7];
  const float* Wv   = (const float*)d_in[8];
  const float* bv   = (const float*)d_in[9];
  const float* Wo   = (const float*)d_in[10];
  const float* bo   = (const float*)d_in[11];

  const size_t SZX = (size_t)MTOT * DMODEL;     // 4,194,304
  const size_t SZW = (size_t)DMODEL * DMODEL;   // 1,048,576
  bf16_t* ws  = (bf16_t*)d_ws;
  bf16_t* Xq  = ws;
  bf16_t* Xk  = Xq + SZX;
  bf16_t* Xv  = Xk + SZX;
  bf16_t* Wqb = Xv + SZX;
  bf16_t* Wkb = Wqb + SZW;
  bf16_t* Wvb = Wkb + SZW;
  bf16_t* Wob = Wvb + SZW;
  bf16_t* Qb  = Wob + SZW;
  bf16_t* Kb  = Qb + SZX;
  bf16_t* Vt  = Kb + SZX;   // (B,H,DK,S)
  bf16_t* Ctx = Vt + SZX;   // total 64 MB of workspace

  auto cvt = [&](const float* s, bf16_t* d, size_t n) {
    int n4 = (int)(n / 4);
    cvt4_kernel<<<(n4 + 255) / 256, 256, 0, stream>>>(s, d, n4);
  };
  cvt(q, Xq, SZX);
  cvt(k, Xk, SZX);
  cvt(v, Xv, SZX);
  cvt(Wq, Wqb, SZW);
  cvt(Wk, Wkb, SZW);
  cvt(Wv, Wvb, SZW);
  cvt(Wo, Wob, SZW);

  dim3 gg(MTOT / 128, DMODEL / 128);  // (32, 8)
  gemm_proj_kernel<<<gg, 256, 0, stream>>>(Xq, Wqb, bq, nullptr, Qb, 1);
  gemm_proj_kernel<<<gg, 256, 0, stream>>>(Xk, Wkb, bk, nullptr, Kb, 1);
  gemm_proj_kernel<<<gg, 256, 0, stream>>>(Xv, Wvb, bv, nullptr, Vt, 2);

  dim3 ga(SEQ / 64, BATCH * NHEAD);   // (32, 32)
  attn_kernel<<<ga, 128, 0, stream>>>(Qb, Kb, Vt, mask, Ctx);

  gemm_proj_kernel<<<gg, 256, 0, stream>>>(Ctx, Wob, bo, (float*)d_out, nullptr, 0);
}
